// CaptionDecoder_68350109548982
// MI455X (gfx1250) — matrix operand order repaired and checked by reading, not compiled, not run on hardware
//
#include <hip/hip_runtime.h>
#include <hip/hip_bf16.h>

// ---------------------------------------------------------------------------
// CDNA5 (gfx1250) caption-decoder: bf16 WMMA GEMMs + fp32 attention/LSTM.
// ---------------------------------------------------------------------------

typedef __bf16 bf16_t;
typedef __attribute__((ext_vector_type(16))) __bf16        v16bf;
typedef __attribute__((ext_vector_type(8)))  float         v8f;
typedef __attribute__((ext_vector_type(4)))  unsigned int  v4u;

#define E_ 512
#define H_ 512
#define V_ 32000
#define B_ 32
#define P_ 196
#define T_ 32

__device__ __forceinline__ bf16_t f2bf(float f) {
  unsigned u = __builtin_bit_cast(unsigned, f);
  u += 0x7fffu + ((u >> 16) & 1u);             // round-to-nearest-even
  unsigned short h = (unsigned short)(u >> 16);
  return __builtin_bit_cast(bf16_t, h);
}
__device__ __forceinline__ float sigmoidf_(float x) {
  return 1.0f / (1.0f + __expf(-x));
}

// ------------------------- generic bf16 WMMA GEMM --------------------------
// C[m][n] = sum_k A[m][k] * W[n][k] + bias[n]
// A: M x K row-major bf16, W: N x K row-major bf16 (i.e. weight as stored).
// One wave32 computes one 16x16 tile via v_wmma_f32_16x16x32_bf16.
__global__ void k_gemm_bf16(const bf16_t* __restrict__ A,
                            const bf16_t* __restrict__ W,
                            const float*  __restrict__ bias,
                            float* __restrict__ C, long long ldc,
                            int M, int N, int K)
{
  const int wave = threadIdx.x >> 5;
  const int lane = threadIdx.x & 31;
  const int mt = M >> 4;
  const int nt = N >> 4;
  int tile = blockIdx.x * (blockDim.x >> 5) + wave;
  if (tile >= mt * nt) return;                 // uniform per wave: EXEC all-1s
  const int m0   = (tile % mt) << 4;
  const int n0   = (tile / mt) << 4;
  const int half = lane >> 4;                  // lane group 0-15 / 16-31
  const int li   = lane & 15;

  // A fragment: row = m0+li; lanes 0-15 hold K 0..7 & 16..23, lanes 16-31
  // hold K 8..15 & 24..31 (ISA 16-bit A layout).
  const bf16_t* pa = A + (size_t)(m0 + li) * K + half * 8;
  // B fragment: col = n0+li; same K split as A (0..7 & 16..23 / 8..15 & 24..31).
  const bf16_t* pb = W + (size_t)(n0 + li) * K + half * 8;

  v8f acc = {};
  for (int k0 = 0; k0 < K; k0 += 32) {
    union { v16bf v; v4u q[2]; } a, b;
    a.q[0] = *(const v4u*)(pa);                // K 0..7   (16B)
    a.q[1] = *(const v4u*)(pa + 16);           // K 16..23 (16B)
    b.q[0] = *(const v4u*)(pb);                // K 0..7   (16B)
    b.q[1] = *(const v4u*)(pb + 16);           // K 16..23 (16B)
    acc = __builtin_amdgcn_wmma_f32_16x16x32_bf16(
        /*neg_a=*/false, a.v, /*neg_b=*/false, b.v,
        /*c_mod=*/(short)0, acc, /*reuse_a=*/false, /*reuse_b=*/false);
    pa += 32; pb += 32;
  }
  // D layout: VGPR r -> row m0 + half*8 + r, col = n0 + li.
  const int   col = n0 + li;
  const float bv  = bias ? bias[col] : 0.0f;
  float* cp = C + (size_t)(m0 + half * 8) * ldc + col;
#pragma unroll
  for (int r = 0; r < 8; ++r) cp[(size_t)r * ldc] = acc[r] + bv;
}

// --------------------------- setup / conversion ----------------------------
__global__ void k_f32_to_bf16(const float* __restrict__ s,
                              bf16_t* __restrict__ d, int n) {
  int i = blockIdx.x * blockDim.x + threadIdx.x;
  if (i < n) d[i] = f2bf(s[i]);
}

// split attn_W (H x (E+H)) into Wf (H x E) and Wh (H x H), bf16
__global__ void k_cvt_attn(const float* __restrict__ attn_W,
                           bf16_t* __restrict__ WfB,
                           bf16_t* __restrict__ WhB) {
  int i = blockIdx.x * blockDim.x + threadIdx.x;
  if (i >= H_ * E_) return;
  int r = i / E_, c = i % E_;
  WfB[i] = f2bf(attn_W[(size_t)r * (E_ + H_) + c]);
  WhB[i] = f2bf(attn_W[(size_t)r * (E_ + H_) + E_ + c]);
}

// concatenated gate weight Wg (4H x (2E + H)) = [W_ih | W_hh], bf16
__global__ void k_build_Wg(const float* __restrict__ W_ih,
                           const float* __restrict__ W_hh,
                           bf16_t* __restrict__ WgB) {
  int i = blockIdx.x * blockDim.x + threadIdx.x;
  const int KC = 2 * E_ + H_;                  // 1536
  if (i >= 4 * H_ * KC) return;
  int r = i / KC, k = i % KC;
  float v = (k < 2 * E_) ? W_ih[(size_t)r * (2 * E_) + k]
                         : W_hh[(size_t)r * H_ + (k - 2 * E_)];
  WgB[i] = f2bf(v);
}

__global__ void k_bsum(const float* __restrict__ b_ih,
                       const float* __restrict__ b_hh,
                       float* __restrict__ bsum) {
  int i = blockIdx.x * blockDim.x + threadIdx.x;
  if (i < 4 * H_) bsum[i] = b_ih[i] + b_hh[i];
}

// mean over P positions: mean[b][e]
__global__ void k_mean(const float* __restrict__ features,
                       float* __restrict__ mean) {
  int b = blockIdx.x;
  for (int e = threadIdx.x; e < E_; e += blockDim.x) {
    float s = 0.0f;
    for (int p = 0; p < P_; ++p)
      s += features[((size_t)b * P_ + p) * E_ + e];
    mean[b * E_ + e] = s * (1.0f / P_);
  }
}

// h0 = mean @ inith_W.T + inith_b ; c0 likewise ; also hB = bf16(h0)
__global__ void k_init_hc(const float* __restrict__ mean,
                          const float* __restrict__ inith_W,
                          const float* __restrict__ inith_b,
                          const float* __restrict__ initc_W,
                          const float* __restrict__ initc_b,
                          float* __restrict__ h, float* __restrict__ c,
                          bf16_t* __restrict__ hB) {
  int i = blockIdx.x * blockDim.x + threadIdx.x;
  if (i >= B_ * H_) return;
  int b = i / H_, j = i % H_;
  const float* mb = mean + b * E_;
  float ah = inith_b[j], ac = initc_b[j];
  const float* wh = inith_W + (size_t)j * E_;
  const float* wc = initc_W + (size_t)j * E_;
  for (int e = 0; e < E_; ++e) { ah += mb[e] * wh[e]; ac += mb[e] * wc[e]; }
  h[i] = ah; c[i] = ac; hB[i] = f2bf(ah);
}

// ------------------------------ per-step work ------------------------------
// attention: scores -> softmax -> context; also assembles x = [emb,ctx,h] bf16
__global__ void k_attn_step(const float* __restrict__ feat_proj,
                            const float* __restrict__ features,
                            const float* __restrict__ hWh,
                            const float* __restrict__ attn_b,
                            const float* __restrict__ v_w,
                            const float* __restrict__ h,
                            const float* __restrict__ embed_W,
                            const int*   __restrict__ captions,
                            bf16_t* __restrict__ xB, int t)
{
  __shared__ float s_hb[H_];
  __shared__ float s_vw[H_];
  __shared__ float s_sc[P_];
  __shared__ float s_inv;
  const int b = blockIdx.x, tid = threadIdx.x;
  const int wave = tid >> 5, lane = tid & 31;

  for (int j = tid; j < H_; j += blockDim.x) {
    s_hb[j] = hWh[b * H_ + j] + attn_b[j];
    s_vw[j] = v_w[j];
  }
  __syncthreads();

  // scores[p] = sum_h v_w[h] * tanh(feat_proj[b,p,h] + hWh[b,h] + attn_b[h])
  for (int p = wave; p < P_; p += 8) {
    const float* fp = feat_proj + ((size_t)b * P_ + p) * H_;
    float s = 0.0f;
    for (int hh = lane; hh < H_; hh += 32)
      s += s_vw[hh] * tanhf(fp[hh] + s_hb[hh]);
    for (int o = 16; o; o >>= 1) s += __shfl_xor(s, o, 32);
    if (lane == 0) s_sc[p] = s;
  }
  __syncthreads();

  // softmax over P (one wave)
  if (tid < 32) {
    float m = -1e30f;
    for (int p = tid; p < P_; p += 32) m = fmaxf(m, s_sc[p]);
    for (int o = 16; o; o >>= 1) m = fmaxf(m, __shfl_xor(m, o, 32));
    float s = 0.0f;
    for (int p = tid; p < P_; p += 32) {
      float e = __expf(s_sc[p] - m); s_sc[p] = e; s += e;
    }
    for (int o = 16; o; o >>= 1) s += __shfl_xor(s, o, 32);
    if (tid == 0) s_inv = 1.0f / s;
  }
  __syncthreads();
  const float inv = s_inv;

  // context + x assembly: x = [emb(E) | context(E) | h(H)]  (K = 1536)
  const int cap = captions[b * T_ + t];
  bf16_t* xb = xB + (size_t)b * (2 * E_ + H_);
  for (int e = tid; e < E_; e += blockDim.x) {
    float acc = 0.0f;
    const float* fb = features + (size_t)b * P_ * E_ + e;
    for (int p = 0; p < P_; ++p) acc += s_sc[p] * fb[(size_t)p * E_];
    acc *= inv;
    xb[e]          = f2bf(embed_W[(size_t)cap * E_ + e]);
    xb[E_ + e]     = f2bf(acc);
    xb[2 * E_ + e] = f2bf(h[b * H_ + e]);
  }
}

// LSTM pointwise update; writes h (f32), c (f32) and hB (bf16 for GEMMs)
__global__ void k_lstm_step(const float* __restrict__ gates,
                            float* __restrict__ h, float* __restrict__ c,
                            bf16_t* __restrict__ hB) {
  int idx = blockIdx.x * blockDim.x + threadIdx.x;
  if (idx >= B_ * H_) return;
  int b = idx / H_, j = idx % H_;
  const float* g = gates + (size_t)b * 4 * H_;
  float i_ = sigmoidf_(g[j]);
  float f_ = sigmoidf_(g[H_ + j]);
  float gg = tanhf(g[2 * H_ + j]);
  float o_ = sigmoidf_(g[3 * H_ + j]);
  float cn = f_ * c[idx] + i_ * gg;
  float hn = o_ * tanhf(cn);
  c[idx] = cn; h[idx] = hn; hB[idx] = f2bf(hn);
}

// ------------------------------- launcher ----------------------------------
extern "C" void kernel_launch(void* const* d_in, const int* in_sizes, int n_in,
                              void* d_out, int out_size, void* d_ws, size_t ws_size,
                              hipStream_t stream) {
  const float* features = (const float*)d_in[0];
  const int*   captions = (const int*)d_in[1];
  const float* embed_W  = (const float*)d_in[2];
  const float* attn_W   = (const float*)d_in[3];
  const float* attn_b   = (const float*)d_in[4];
  const float* v_w      = (const float*)d_in[5];
  const float* W_ih     = (const float*)d_in[6];
  const float* W_hh     = (const float*)d_in[7];
  const float* b_ih     = (const float*)d_in[8];
  const float* b_hh     = (const float*)d_in[9];
  const float* lin_W    = (const float*)d_in[10];
  const float* lin_b    = (const float*)d_in[11];
  const float* inith_W  = (const float*)d_in[12];
  const float* inith_b  = (const float*)d_in[13];
  const float* initc_W  = (const float*)d_in[14];
  const float* initc_b  = (const float*)d_in[15];
  float* out = (float*)d_out;

  const int KC = 2 * E_ + H_;                      // 1536
  char* p = (char*)d_ws;
  auto alloc = [&](size_t bytes) -> char* {
    char* r = p; p += (bytes + 255) & ~(size_t)255; return r;
  };
  bf16_t* featB   = (bf16_t*)alloc((size_t)B_ * P_ * E_ * 2);
  bf16_t* WfB     = (bf16_t*)alloc((size_t)H_ * E_ * 2);
  bf16_t* WhB     = (bf16_t*)alloc((size_t)H_ * H_ * 2);
  bf16_t* WgB     = (bf16_t*)alloc((size_t)4 * H_ * KC * 2);
  bf16_t* linWB   = (bf16_t*)alloc((size_t)V_ * H_ * 2);
  float*  featprj = (float*) alloc((size_t)B_ * P_ * H_ * 4);
  float*  meanf   = (float*) alloc((size_t)B_ * E_ * 4);
  float*  h       = (float*) alloc((size_t)B_ * H_ * 4);
  float*  c       = (float*) alloc((size_t)B_ * H_ * 4);
  bf16_t* hB      = (bf16_t*)alloc((size_t)B_ * H_ * 2);
  float*  hWh     = (float*) alloc((size_t)B_ * H_ * 4);
  bf16_t* xB      = (bf16_t*)alloc((size_t)B_ * KC * 2);
  float*  gates   = (float*) alloc((size_t)B_ * 4 * H_ * 4);
  float*  bsum    = (float*) alloc((size_t)4 * H_ * 4);

  auto gemm = [&](const bf16_t* A, const bf16_t* W, const float* bias,
                  float* C, long long ldc, int M, int N, int K) {
    int tiles  = (M / 16) * (N / 16);
    int blocks = (tiles + 7) / 8;                  // 8 waves / block
    hipLaunchKernelGGL(k_gemm_bf16, dim3(blocks), dim3(256), 0, stream,
                       A, W, bias, C, ldc, M, N, K);
  };
  auto cvt = [&](const float* s, bf16_t* d, int n) {
    hipLaunchKernelGGL(k_f32_to_bf16, dim3((n + 255) / 256), dim3(256), 0,
                       stream, s, d, n);
  };

  // one-time setup
  hipLaunchKernelGGL(k_mean, dim3(B_), dim3(256), 0, stream, features, meanf);
  hipLaunchKernelGGL(k_init_hc, dim3((B_ * H_ + 255) / 256), dim3(256), 0,
                     stream, meanf, inith_W, inith_b, initc_W, initc_b,
                     h, c, hB);
  cvt(features, featB, B_ * P_ * E_);
  cvt(lin_W, linWB, V_ * H_);
  hipLaunchKernelGGL(k_cvt_attn, dim3((H_ * E_ + 255) / 256), dim3(256), 0,
                     stream, attn_W, WfB, WhB);
  hipLaunchKernelGGL(k_build_Wg, dim3((4 * H_ * KC + 255) / 256), dim3(256), 0,
                     stream, W_ih, W_hh, WgB);
  hipLaunchKernelGGL(k_bsum, dim3((4 * H_ + 255) / 256), dim3(256), 0,
                     stream, b_ih, b_hh, bsum);

  // feat_proj = features @ Wf.T   (6272 x 512 x 512)
  gemm(featB, WfB, nullptr, featprj, H_, B_ * P_, H_, E_);

  // T sequential decode steps
  for (int t = 0; t < T_; ++t) {
    // hWh = h @ Wh.T
    gemm(hB, WhB, nullptr, hWh, H_, B_, H_, H_);
    // attention + assemble x = [emb | context | h]
    hipLaunchKernelGGL(k_attn_step, dim3(B_), dim3(256), 0, stream,
                       featprj, features, hWh, attn_b, v_w, h,
                       embed_W, captions, xB, t);
    // gates = x @ [W_ih|W_hh].T + (b_ih + b_hh)
    gemm(xB, WgB, bsum, gates, 4 * H_, B_, 4 * H_, KC);
    // LSTM pointwise
    hipLaunchKernelGGL(k_lstm_step, dim3((B_ * H_ + 255) / 256), dim3(256), 0,
                       stream, gates, h, c, hB);
    // logits[:, t, :] = h_new @ lin_W.T + lin_b
    gemm(hB, linWB, lin_b, out + (size_t)t * V_, (long long)T_ * V_,
         B_, V_, H_);
  }
}
